// KalmanVAE_toeplitz_23819888624215
// MI455X (gfx1250) — matrix operand order repaired and checked by reading, not compiled, not run on hardware
//
#include <hip/hip_runtime.h>
#include <hip/hip_bf16.h>

typedef __bf16 bf16_t;
typedef __attribute__((ext_vector_type(16))) __bf16 v16bf;
typedef __attribute__((ext_vector_type(8)))  float  v8f;

#define BM 64
#define BN 32
#define BK 32
#define XPAD 8   // LDS row pad (bf16 elems): stride 40 elems = 80B, 16B-aligned

__device__ __forceinline__ int imind(int a, int b) { return a < b ? a : b; }

// ---------------------------------------------------------------------------
// Fused GEMM: Y[M,N] = act( X[M,K](bf16, ldx mult of 8) @ W[N,K]^T(f32) + b )
// gridDim.z = units (per-unit strides inside; z==0 => no offset).
// act: 0=none, 1=relu, 2=clip(-4,4). Writes bf16 and/or f32.
// Requirements: M multiple of BM; ldx multiple of 8; X base 16B aligned.
// Inner loop: 1x B128 global load (X), 4x dword loads (W, clamped row,
// cndmask zero-fill), barrier, ds_load fragments, v_wmma_f32_16x16x32_bf16.
// ---------------------------------------------------------------------------
__global__ __launch_bounds__(256)
void gemm_bias_act(const bf16_t* __restrict__ X, int ldx,
                   const float* __restrict__ W,
                   const float* __restrict__ bias,
                   bf16_t* __restrict__ Ybf, float* __restrict__ Yf, int ldy,
                   int M, int N, int K, int act)
{
    __shared__ bf16_t lx[BM][BK + XPAD];   // A tile [m][k]
    __shared__ bf16_t lb[BK][BN + XPAD];   // B tile [k][n] (= W^T)

    const int u = blockIdx.z;
    X    += (size_t)u * M * ldx;
    W    += (size_t)u * N * K;
    bias += (size_t)u * N;
    if (Ybf) Ybf += (size_t)u * M * ldy;
    if (Yf)  Yf  += (size_t)u * M * ldy;

    const int tid  = threadIdx.x;
    const int lane = tid & 31;
    const int wid  = tid >> 5;
    const int mw   = (wid & 3) * 16;
    const int nw   = (wid >> 2) * 16;
    const int m0   = blockIdx.x * BM;
    const int n0   = blockIdx.y * BN;

    // staging slots
    const int xrow = tid >> 2, xkk = (tid & 3) * 8;        // X: 8 bf16 (16B)
    const int wn   = tid >> 3, wkk = (tid & 7) * 4;        // W: 4 f32
    const int gm   = m0 + xrow;                            // always < M
    const int gn   = n0 + wn;
    const bool wok = (gn < N);
    const float* wrow = W + (size_t)imind(gn, N - 1) * K;  // clamped, always valid

    // fragment read positions
    const int arow = mw + (lane & 15);
    const int akh  = (lane >> 4) * 8;

    v8f acc = {};
    const int Kfull = K & ~(BK - 1);

    for (int k0 = 0; k0 < Kfull; k0 += BK) {
        // X tile: one aligned 16B load, no guards
        *(uint4*)&lx[xrow][xkk] = *(const uint4*)(X + (size_t)gm * ldx + k0 + xkk);
        // W tile (transposed into LDS, f32 -> bf16): branch-free zero-fill
        {
            const float* wp = wrow + k0 + wkk;
            float w0 = wp[0], w1 = wp[1], w2 = wp[2], w3 = wp[3];
            if (!wok) { w0 = w1 = w2 = w3 = 0.f; }   // cndmask, not exec branch
            lb[wkk + 0][wn] = (bf16_t)w0;
            lb[wkk + 1][wn] = (bf16_t)w1;
            lb[wkk + 2][wn] = (bf16_t)w2;
            lb[wkk + 3][wn] = (bf16_t)w3;
        }
        __syncthreads();
        v16bf a, b;
        #pragma unroll
        for (int i = 0; i < 8; ++i) {
            a[i]     = lx[arow][akh + i];
            a[8 + i] = lx[arow][16 + akh + i];
        }
        #pragma unroll
        for (int i = 0; i < 8; ++i) {
            b[i]     = lb[akh + i][nw + (lane & 15)];
            b[8 + i] = lb[16 + akh + i][nw + (lane & 15)];
        }
        acc = __builtin_amdgcn_wmma_f32_16x16x32_bf16(
                  false, a, false, b, (short)0, acc, false, false);
        __syncthreads();
    }

    if (Kfull < K) {   // single guarded tail tile (uniform branch, EXEC intact)
        #pragma unroll
        for (int j = 0; j < 8; ++j) {
            const int gk = Kfull + xkk + j;
            const bf16_t v = X[(size_t)gm * ldx + imind(gk, K - 1)];
            lx[xrow][xkk + j] = (gk < K) ? v : (bf16_t)0.0f;
        }
        #pragma unroll
        for (int j = 0; j < 4; ++j) {
            const int gk = Kfull + wkk + j;
            const float wv = wrow[imind(gk, K - 1)];
            lb[wkk + j][wn] = (wok && gk < K) ? (bf16_t)wv : (bf16_t)0.0f;
        }
        __syncthreads();
        v16bf a, b;
        #pragma unroll
        for (int i = 0; i < 8; ++i) {
            a[i]     = lx[arow][akh + i];
            a[8 + i] = lx[arow][16 + akh + i];
        }
        #pragma unroll
        for (int i = 0; i < 8; ++i) {
            b[i]     = lb[akh + i][nw + (lane & 15)];
            b[8 + i] = lb[16 + akh + i][nw + (lane & 15)];
        }
        acc = __builtin_amdgcn_wmma_f32_16x16x32_bf16(
                  false, a, false, b, (short)0, acc, false, false);
        __syncthreads();
    }

    // epilogue: C/D layout — VGPR r: lanes 0-15 -> M=r, lanes 16-31 -> M=8+r
    const int ncol  = n0 + nw + (lane & 15);
    const int mbase = m0 + mw + ((lane >> 4) << 3);
    if (ncol < N) {
        const float bv = bias[ncol];
        #pragma unroll
        for (int r = 0; r < 8; ++r) {
            const int gm2 = mbase + r;
            float v = acc[r] + bv;
            if (act == 1)      v = fmaxf(v, 0.0f);
            else if (act == 2) v = fminf(fmaxf(v, -4.0f), 4.0f);
            if (Ybf) Ybf[(size_t)gm2 * ldy + ncol] = (bf16_t)v;
            if (Yf)  Yf [(size_t)gm2 * ldy + ncol] = v;
        }
    }
}

// ---------------------------------------------------------------------------
// BatchNorm stats: per-column mean/var over M rows; one block = 32 columns.
// ---------------------------------------------------------------------------
__global__ __launch_bounds__(256)
void bn_stats(const bf16_t* __restrict__ Y, int ldy, int M, int N,
              float* __restrict__ stats)
{
    const int u = blockIdx.z;
    Y     += (size_t)u * M * ldy;
    stats += (size_t)u * N * 2;
    const int c = blockIdx.x * 32 + (threadIdx.x & 31);
    const int r = threadIdx.x >> 5;
    float s = 0.f, ss = 0.f;
    if (c < N)
        for (int m = r; m < M; m += 8) {
            const float v = (float)Y[(size_t)m * ldy + c];
            s += v; ss += v * v;
        }
    __shared__ float l1[8][32], l2[8][32];
    l1[r][threadIdx.x & 31] = s;
    l2[r][threadIdx.x & 31] = ss;
    __syncthreads();
    if (r == 0 && c < N) {
        #pragma unroll
        for (int i = 1; i < 8; ++i) {
            s  += l1[i][threadIdx.x & 31];
            ss += l2[i][threadIdx.x & 31];
        }
        const float mean = s / M;
        stats[c * 2]     = mean;
        stats[c * 2 + 1] = ss / M - mean * mean;
    }
}

__global__ __launch_bounds__(256)
void bn_apply(const bf16_t* __restrict__ src, int lds_,
              bf16_t* __restrict__ dst, int ldd,
              const float* __restrict__ stats, int M, int N, float eps)
{
    const int u = blockIdx.z;
    src   += (size_t)u * M * lds_;
    dst   += (size_t)u * M * ldd;
    stats += (size_t)u * N * 2;
    const int idx = blockIdx.x * 256 + threadIdx.x;
    if (idx >= M * N) return;
    const int m = idx / N, n = idx % N;
    const float v = ((float)src[(size_t)m * lds_ + n] - stats[n * 2]) *
                    rsqrtf(stats[n * 2 + 1] + eps);
    dst[(size_t)m * ldd + n] = (bf16_t)v;
}

// ---------------------------------------------------------------------------
// Conv path (3x3 on width-1 image degenerates to 1-D conv over height).
// ---------------------------------------------------------------------------
__global__ __launch_bounds__(256)
void conv1_relu(const float* __restrict__ x, const float* __restrict__ w1,
                const float* __restrict__ b1, float* __restrict__ y1)
{
    const int u = blockIdx.z;
    const int idx = blockIdx.x * 256 + threadIdx.x;   // b*128 + c*32 + h
    if (idx >= 1024 * 128) return;
    const int b = idx >> 7, c = (idx >> 5) & 3, h = idx & 31;
    float acc = b1[u * 4 + c];
    #pragma unroll
    for (int ci = 0; ci < 2; ++ci)
        #pragma unroll
        for (int dh = -1; dh <= 1; ++dh) {
            const int hh = h + dh;
            if (hh < 0 || hh >= 32) continue;
            acc += x[(((size_t)b * 2 + ci) * 32 + hh) * 16 + u] *
                   w1[(((u * 4 + c) * 2 + ci) * 3 + (dh + 1)) * 3 + 1];
        }
    y1[((size_t)u * 1024 + b) * 128 + c * 32 + h] = fmaxf(acc, 0.f);
}

__global__ __launch_bounds__(256)
void conv_stats(const float* __restrict__ y, int C, float* __restrict__ stats)
{
    const int u = blockIdx.z, c = blockIdx.x;
    const float* p = y + (size_t)u * 1024 * C * 32;
    float s = 0.f, ss = 0.f;
    for (int i = threadIdx.x; i < 1024 * 32; i += 256) {
        const int b = i >> 5, h = i & 31;
        const float v = p[((size_t)b * C + c) * 32 + h];
        s += v; ss += v * v;
    }
    __shared__ float r1[256], r2[256];
    r1[threadIdx.x] = s; r2[threadIdx.x] = ss;
    __syncthreads();
    for (int o = 128; o > 0; o >>= 1) {
        if (threadIdx.x < o) {
            r1[threadIdx.x] += r1[threadIdx.x + o];
            r2[threadIdx.x] += r2[threadIdx.x + o];
        }
        __syncthreads();
    }
    if (threadIdx.x == 0) {
        const float inv = 1.0f / 32768.0f;
        const float mean = r1[0] * inv;
        stats[(u * C + c) * 2]     = mean;
        stats[(u * C + c) * 2 + 1] = r2[0] * inv - mean * mean;
    }
}

__global__ __launch_bounds__(256)
void conv2_relu(const float* __restrict__ y1, const float* __restrict__ st1,
                const float* __restrict__ w2, const float* __restrict__ b2,
                float* __restrict__ y2)
{
    const int u = blockIdx.z;
    const int idx = blockIdx.x * 256 + threadIdx.x;   // b*256 + c*32 + h
    if (idx >= 1024 * 256) return;
    const int b = idx >> 8, c = (idx >> 5) & 7, h = idx & 31;
    float acc = b2[u * 8 + c];
    #pragma unroll
    for (int ci = 0; ci < 4; ++ci) {
        const float v = (y1[((size_t)u * 1024 + b) * 128 + ci * 32 + h] -
                         st1[(u * 4 + ci) * 2]) *
                        rsqrtf(st1[(u * 4 + ci) * 2 + 1] + 1e-5f);
        acc += v * w2[(u * 8 + c) * 4 + ci];
    }
    y2[((size_t)u * 1024 + b) * 256 + c * 32 + h] = fmaxf(acc, 0.f);
}

__global__ __launch_bounds__(256)
void conv_bn_obuf(const float* __restrict__ y2, const float* __restrict__ st2,
                  bf16_t* __restrict__ obuf)
{
    const int u = blockIdx.z;
    const int idx = blockIdx.x * 256 + threadIdx.x;   // b*256 + col
    if (idx >= 1024 * 256) return;
    const int b = idx >> 8, col = idx & 255, c = col >> 5;
    const float v = (y2[((size_t)u * 1024 + b) * 256 + col] - st2[(u * 8 + c) * 2]) *
                    rsqrtf(st2[(u * 8 + c) * 2 + 1] + 1e-5f);
    obuf[((size_t)u * 1024 + b) * 360 + col] = (bf16_t)v;
}

__global__ __launch_bounds__(256)
void zero_bf16(bf16_t* p, int n)
{
    const int i = blockIdx.x * 256 + threadIdx.x;
    if (i < n) p[i] = (bf16_t)0.0f;
}

__global__ __launch_bounds__(256)
void reparam(const float* __restrict__ mu, const float* __restrict__ lv,
             const float* __restrict__ ep, float* __restrict__ zf,
             bf16_t* __restrict__ zcur, bf16_t* __restrict__ zseq,
             bf16_t* __restrict__ zin_next)
{
    const int i = blockIdx.x * 256 + threadIdx.x;
    if (i >= 1024 * 32) return;
    const float z = mu[i] + ep[i] * expf(0.5f * lv[i]);
    zf[i] = z;
    const bf16_t zb = (bf16_t)z;
    zcur[i] = zb;
    zseq[i] = zb;
    if (zin_next) zin_next[i] = zb;
}

__global__ __launch_bounds__(256)
void toeplitz(const float* __restrict__ araw, const float* __restrict__ rraw,
              float* __restrict__ Bm, float* __restrict__ Cm)
{
    const int ub = blockIdx.x;          // u*1024 + b
    __shared__ float Ar[32], Ai[32];
    const int tid = threadIdx.x;
    if (tid < 64) {
        const float a0 = fminf(expf(araw[ub]), 500.0f);
        if (tid == 0)      { Ar[0] = a0; Ai[0] = 0.f; }
        else if (tid < 32)   Ar[tid] = 0.022f * a0 * tanhf(rraw[(size_t)ub * 62 + tid - 1]);
        else if (tid > 32) { const int t = tid - 32;
                             Ai[t] = 0.022f * a0 * tanhf(rraw[(size_t)ub * 62 + 30 + t]); }
    }
    __syncthreads();
    const size_t base = (size_t)ub * 1024;
    for (int e = tid; e < 1024; e += 256) {
        const int i = e >> 5, j = e & 31, d = i - j;
        float br = 0.f, bi = 0.f, cr = 0.f, ci = 0.f;
        if (d >= 0) { br = Ar[d]; bi = Ai[d]; }
        if (d >= 1) { const int g = 32 - d; cr = Ar[g]; ci = -Ai[g]; }
        const size_t o = (base + e) * 2;
        Bm[o] = br; Bm[o + 1] = bi;
        Cm[o] = cr; Cm[o + 1] = ci;
    }
}

// ---------------------------------------------------------------------------
// Host orchestration
// ---------------------------------------------------------------------------
static void gemm(hipStream_t s, const bf16_t* X, int ldx, const float* W,
                 const float* bias, bf16_t* Ybf, float* Yf, int ldy,
                 int M, int N, int K, int act, int nz)
{
    dim3 g((M + BM - 1) / BM, (N + BN - 1) / BN, nz);
    gemm_bias_act<<<g, 256, 0, s>>>(X, ldx, W, bias, Ybf, Yf, ldy, M, N, K, act);
}

static void bnorm(hipStream_t s, const bf16_t* src, int lds_, bf16_t* dst, int ldd,
                  float* stats, int M, int N, float eps, int nz)
{
    bn_stats<<<dim3((N + 31) / 32, 1, nz), 256, 0, s>>>(src, lds_, M, N, stats);
    bn_apply<<<dim3((M * N + 255) / 256, 1, nz), 256, 0, s>>>(src, lds_, dst, ldd,
                                                              stats, M, N, eps);
}

enum {
    I_e_c1w = 0, I_e_c1b, I_e_c2w, I_e_c2b, I_e_z1w, I_e_z1b, I_e_z2w, I_e_z2b,
    I_e_n1w, I_e_n1b, I_e_n2w, I_e_n2b, I_e_m1w, I_e_m1b, I_e_m2w, I_e_m2b,
    I_e_m3w, I_e_m3b, I_e_v1w, I_e_v1b, I_e_v2w, I_e_v2b, I_e_v3w, I_e_v3b,
    I_d_n1w, I_d_n1b, I_d_n2w, I_d_n2b, I_d_m1w, I_d_m1b, I_d_m2w, I_d_m2b,
    I_d_a1w, I_d_a1b, I_d_a2w, I_d_a2b, I_d_r1w, I_d_r1b, I_d_r2w, I_d_r2b,
    I_p_n1w, I_p_n1b, I_p_n2w, I_p_n2b, I_p_m1w, I_p_m1b, I_p_m2w, I_p_m2b,
    I_p_l1w, I_p_l1b, I_p_l2w, I_p_l2b
};

extern "C" void kernel_launch(void* const* d_in, const int* in_sizes, int n_in,
                              void* d_out, int out_size, void* d_ws, size_t ws_size,
                              hipStream_t stream)
{
    const float* x  = (const float*)d_in[0];
    const float* ep = (const float*)d_in[1];
    const float* P[52];
    for (int i = 0; i < 52; ++i) P[i] = (const float*)d_in[2 + i];

    // ---- workspace carve-up (bump allocator, 256B aligned) ----
    // All bf16 buffers use leading dims that are multiples of 8 (16B rows).
    char* w = (char*)d_ws;
    auto alloc = [&](size_t bytes) -> char* {
        char* p = w; w += (bytes + 255) & ~(size_t)255; return p;
    };
    float*  STATS = (float*)alloc(16 * 400 * 2 * sizeof(float));
    bf16_t* OBUF  = (bf16_t*)alloc((size_t)16 * 1024 * 360 * 2);  // ld 360
    bf16_t* ZCUR  = (bf16_t*)alloc(1024 * 32 * 2);
    bf16_t* EA    = (bf16_t*)alloc(1024 * 400 * 2);
    bf16_t* EB    = (bf16_t*)alloc(1024 * 400 * 2);
    bf16_t* EC    = (bf16_t*)alloc(1024 * 200 * 2);
    bf16_t* ED    = (bf16_t*)alloc(1024 * 104 * 2);               // ld 104
    bf16_t* EE    = (bf16_t*)alloc(1024 * 104 * 2);
    bf16_t* ZSEQ  = (bf16_t*)alloc((size_t)16384 * 32 * 2);
    bf16_t* ZIN   = (bf16_t*)alloc((size_t)16384 * 32 * 2);
    float*  ARAW  = (float*)alloc(16384 * sizeof(float));
    float*  RRAW  = (float*)alloc((size_t)16384 * 62 * sizeof(float));
    char*   BIG   = alloc(32768000);                // conv scratch / decoder bufs
    float*  Y1    = (float*)BIG;                    // [16,1024,4,32] (dead after conv)
    float*  Y2    = (float*)(BIG + 8388608);        // [16,1024,8,32]
    bf16_t* D1    = (bf16_t*)BIG;                   // [16,1024,<=400]
    bf16_t* D2    = (bf16_t*)(BIG + 13107200);      // [16,1024,<=400]
    bf16_t* D3    = (bf16_t*)(BIG + 26214400);      // [16,1024,200]

    // ---- output regions (return-order concat; complex64 -> interleaved f32) ----
    float* out  = (float*)d_out;
    float* O_MU = out;                 // [16,1024,2,32,1]
    float* O_B  = out + 1048576;       // [16,1024,32,32] complex
    float* O_C  = out + 34603008;
    float* O_MQ = out + 68157440;      // [16,1024,32]
    float* O_LQ = out + 68681728;
    float* O_MP = out + 69206016;
    float* O_LP = out + 69730304;
    float* O_Z  = out + 70254592;

    // ---- conv/xf path: all 16 units in parallel ----
    conv1_relu<<<dim3(512, 1, 16), 256, 0, stream>>>(x, P[I_e_c1w], P[I_e_c1b], Y1);
    conv_stats<<<dim3(4, 1, 16), 256, 0, stream>>>(Y1, 4, STATS);
    conv2_relu<<<dim3(1024, 1, 16), 256, 0, stream>>>(Y1, STATS, P[I_e_c2w], P[I_e_c2b], Y2);
    conv_stats<<<dim3(8, 1, 16), 256, 0, stream>>>(Y2, 8, STATS);
    conv_bn_obuf<<<dim3(1024, 1, 16), 256, 0, stream>>>(Y2, STATS, OBUF);

    zero_bf16<<<128, 256, 0, stream>>>(ZCUR, 1024 * 32);
    zero_bf16<<<128, 256, 0, stream>>>(ZIN, 1024 * 32);      // z_in for unit 0

    // ---- encoder scan over units (z-chain is the only sequential dep) ----
    for (int u = 0; u < 16; ++u) {
        gemm(stream, ZCUR, 32, P[I_e_z1w] + u * 100 * 32, P[I_e_z1b] + u * 100,
             ED, nullptr, 104, 1024, 100, 32, 1, 1);
        bnorm(stream, ED, 104, ED, 104, STATS, 1024, 100, 0.02f, 1);
        gemm(stream, ED, 104, P[I_e_z2w] + u * 100 * 100, P[I_e_z2b] + u * 100,
             EE, nullptr, 104, 1024, 100, 100, 1, 1);
        bnorm(stream, EE, 104, OBUF + (size_t)u * 1024 * 360 + 256, 360,
              STATS, 1024, 100, 0.02f, 1);                    // zf -> concat slot

        const bf16_t* OU = OBUF + (size_t)u * 1024 * 360;
        gemm(stream, OU, 360, P[I_e_n1w] + u * 400 * 356, P[I_e_n1b] + u * 400,
             EA, nullptr, 400, 1024, 400, 356, 1, 1);
        bnorm(stream, EA, 400, EA, 400, STATS, 1024, 400, 1e-5f, 1);
        gemm(stream, EA, 400, P[I_e_n2w] + u * 400 * 400, P[I_e_n2b] + u * 400,
             EB, nullptr, 400, 1024, 400, 400, 1, 1);
        bnorm(stream, EB, 400, EB, 400, STATS, 1024, 400, 1e-5f, 1);

        gemm(stream, EB, 400, P[I_e_m1w] + u * 200 * 400, P[I_e_m1b] + u * 200,
             EC, nullptr, 200, 1024, 200, 400, 1, 1);
        bnorm(stream, EC, 200, EC, 200, STATS, 1024, 200, 1e-5f, 1);
        gemm(stream, EC, 200, P[I_e_m2w] + u * 100 * 200, P[I_e_m2b] + u * 100,
             ED, nullptr, 104, 1024, 100, 200, 1, 1);
        bnorm(stream, ED, 104, ED, 104, STATS, 1024, 100, 1e-5f, 1);
        gemm(stream, ED, 104, P[I_e_m3w] + u * 32 * 100, P[I_e_m3b] + u * 32,
             nullptr, O_MQ + (size_t)u * 32768, 32, 1024, 32, 100, 0, 1);

        gemm(stream, EB, 400, P[I_e_v1w] + u * 200 * 400, P[I_e_v1b] + u * 200,
             EC, nullptr, 200, 1024, 200, 400, 1, 1);
        bnorm(stream, EC, 200, EC, 200, STATS, 1024, 200, 1e-5f, 1);
        gemm(stream, EC, 200, P[I_e_v2w] + u * 100 * 200, P[I_e_v2b] + u * 100,
             ED, nullptr, 104, 1024, 100, 200, 1, 1);
        bnorm(stream, ED, 104, ED, 104, STATS, 1024, 100, 1e-5f, 1);
        gemm(stream, ED, 104, P[I_e_v3w] + u * 32 * 100, P[I_e_v3b] + u * 32,
             nullptr, O_LQ + (size_t)u * 32768, 32, 1024, 32, 100, 0, 1);

        reparam<<<128, 256, 0, stream>>>(
            O_MQ + (size_t)u * 32768, O_LQ + (size_t)u * 32768, ep + (size_t)u * 32768,
            O_Z + (size_t)u * 32768, ZCUR, ZSEQ + (size_t)u * 32768,
            (u < 15) ? ZIN + (size_t)(u + 1) * 32768 : nullptr);
    }

    // ---- decoder: all 16 units batched (gridDim.z=16, per-unit weights) ----
    gemm(stream, ZSEQ, 32, P[I_d_n1w], P[I_d_n1b], D3, nullptr, 200, 1024, 200, 32, 1, 16);
    bnorm(stream, D3, 200, D3, 200, STATS, 1024, 200, 1e-5f, 16);
    gemm(stream, D3, 200, P[I_d_n2w], P[I_d_n2b], D1, nullptr, 400, 1024, 400, 200, 1, 16);
    bnorm(stream, D1, 400, D1, 400, STATS, 1024, 400, 1e-5f, 16);
    gemm(stream, D1, 400, P[I_d_m1w], P[I_d_m1b], D2, nullptr, 400, 1024, 400, 400, 1, 16);
    bnorm(stream, D2, 400, D2, 400, STATS, 1024, 400, 1e-5f, 16);
    gemm(stream, D2, 400, P[I_d_m2w], P[I_d_m2b], nullptr, O_MU, 64, 1024, 64, 400, 0, 16);
    gemm(stream, D1, 400, P[I_d_a1w], P[I_d_a1b], D2, nullptr, 152, 1024, 150, 400, 1, 16);
    bnorm(stream, D2, 152, D2, 152, STATS, 1024, 150, 1e-5f, 16);
    gemm(stream, D2, 152, P[I_d_a2w], P[I_d_a2b], nullptr, ARAW, 1, 1024, 1, 150, 0, 16);
    gemm(stream, D1, 400, P[I_d_r1w], P[I_d_r1b], D2, nullptr, 304, 1024, 300, 400, 1, 16);
    bnorm(stream, D2, 304, D2, 304, STATS, 1024, 300, 1e-5f, 16);
    gemm(stream, D2, 304, P[I_d_r2w], P[I_d_r2b], nullptr, RRAW, 62, 1024, 62, 300, 0, 16);
    toeplitz<<<16384, 256, 0, stream>>>(ARAW, RRAW, O_B, O_C);

    // ---- prior: all 16 units batched on z_in ----
    gemm(stream, ZIN, 32, P[I_p_n1w], P[I_p_n1b], D3, nullptr, 200, 1024, 200, 32, 1, 16);
    bnorm(stream, D3, 200, D3, 200, STATS, 1024, 200, 0.02f, 16);
    gemm(stream, D3, 200, P[I_p_n2w], P[I_p_n2b], D1, nullptr, 200, 1024, 200, 200, 1, 16);
    bnorm(stream, D1, 200, D1, 200, STATS, 1024, 200, 0.02f, 16);
    gemm(stream, D1, 200, P[I_p_m1w], P[I_p_m1b], D2, nullptr, 104, 1024, 100, 200, 1, 16);
    bnorm(stream, D2, 104, D2, 104, STATS, 1024, 100, 1e-5f, 16);
    gemm(stream, D2, 104, P[I_p_m2w], P[I_p_m2b], nullptr, O_MP, 32, 1024, 32, 100, 0, 16);
    gemm(stream, D1, 200, P[I_p_l1w], P[I_p_l1b], D2, nullptr, 104, 1024, 100, 200, 1, 16);
    bnorm(stream, D2, 104, D2, 104, STATS, 1024, 100, 1e-5f, 16);
    gemm(stream, D2, 104, P[I_p_l2w], P[I_p_l2b], nullptr, O_LP, 32, 1024, 32, 100, 2, 16);

    (void)in_sizes; (void)n_in; (void)out_size; (void)ws_size;
}
